// Attention_Fusion_Bimodal_LNSpecific_Sum_39092792328569
// MI455X (gfx1250) — compile-verified, matrix operand order checked by reading, not run on hardware
//
#include <hip/hip_runtime.h>
#include <hip/hip_bf16.h>

#define DEV __device__ __forceinline__

typedef __attribute__((ext_vector_type(16))) __bf16 v16bf;
typedef __attribute__((ext_vector_type(8)))  float  v8f;
typedef __attribute__((ext_vector_type(4)))  unsigned int v4u;
typedef __attribute__((ext_vector_type(8)))  int v8i;
typedef __attribute__((ext_vector_type(4)))  int v4i;

union AFrag { v16bf v; unsigned short u[16]; };

DEV unsigned short f2bf(float f) {
  unsigned int u = __float_as_uint(f);
  u += 0x7FFFu + ((u >> 16) & 1u);          // round-to-nearest-even
  return (unsigned short)(u >> 16);
}

DEV v8f wmma_bf16(v16bf a, v16bf b, v8f c) {
  // (neg_a, A, neg_b, B, c_mod, C, reuse_a, reuse_b)
  return __builtin_amdgcn_wmma_f32_16x16x32_bf16(false, a, false, b, (short)0, c, false, false);
}

#if __has_builtin(__builtin_amdgcn_tensor_load_to_lds)
#define HAS_TDM 1
#endif

DEV void wait_tensorcnt0() {
#if __has_builtin(__builtin_amdgcn_s_wait_tensorcnt)
  __builtin_amdgcn_s_wait_tensorcnt(0);
#else
  asm volatile("s_wait_tensorcnt 0x0" ::: "memory");
#endif
}

// ---------------------------------------------------------------------------
// WMMA GEMM:  C[M,N] = A[M,K] (f32) * Bw[N,K]^T (bf16) + bias + res, opt ReLU
// Block tile 128x64, K-step 32. 4 waves; each wave owns a 32x64 strip:
// 2 A-frags x 4 B-frags -> 8 WMMAs per K-step.
// B tile staged by the Tensor Data Mover (one descriptor per K-step, using
// D# padding to produce the LDS row stride); falls back to uint4 copies.
// ---------------------------------------------------------------------------
#define BM 128
#define BN 64
#define BK 32
#define LDA 40
#define LDB 40

__global__ __launch_bounds__(128) void k_gemm(
    const float* __restrict__ A, const unsigned short* __restrict__ Bw,
    const float* __restrict__ bias, const float* __restrict__ res,
    float* __restrict__ Cout, int M, int K, int N, int relu)
{
  __shared__ __align__(16) unsigned short As[BM * LDA];
  __shared__ __align__(16) unsigned short Bs[BN * LDB];
  const int tid  = threadIdx.x;
  const int lane = tid & 31;
  const int wm   = tid >> 5;    // wave id = M sub-strip (0..3)
  const int m0 = blockIdx.y * BM;
  const int n0 = blockIdx.x * BN;

  const v8f vz = {0.f,0.f,0.f,0.f,0.f,0.f,0.f,0.f};
  v8f c[2][4] = {{vz, vz, vz, vz}, {vz, vz, vz, vz}};

  for (int kt = 0; kt < K; kt += BK) {
    // ---- stage A (f32 -> bf16) into LDS, 16B global loads / 8B LDS stores
    for (int i = tid; i < (BM * BK) / 4; i += 128) {
      int r = i >> 3, q = i & 7;
      float4 f = *(const float4*)&A[(size_t)(m0 + r) * K + kt + q * 4];
      ushort4 hv;
      hv.x = f2bf(f.x); hv.y = f2bf(f.y); hv.z = f2bf(f.z); hv.w = f2bf(f.w);
      *(ushort4*)&As[r * LDA + q * 4] = hv;
    }

    // ---- stage B (bf16 [N][K]) into LDS
#ifdef HAS_TDM
    if (wm == 0) {
      unsigned long long ga =
          (unsigned long long)(const void*)Bw + (((size_t)n0 * K + kt) << 1);
      unsigned ldsa = (unsigned)(size_t)(const void*)Bs;
      v4u g0;
      g0.x = 1u;                                   // count=1, user descriptor
      g0.y = ldsa;                                 // lds_addr
      g0.z = (unsigned)ga;                         // global_addr[31:0]
      g0.w = ((unsigned)(ga >> 32) & 0x01FFFFFFu)  // global_addr[56:32]
             | (2u << 30);                         // type=2 (image)
      v8i g1;
      g1[0] = (1 << 16)        // data_size = 2 bytes
            | (1 << 20)        // pad_enable
            | (3 << 22)        // pad_interval: 16 dwords (one 32-elem row)
            | (3 << 25);       // pad_amount: 4 dwords (8 ushorts -> stride 40)
      g1[1] = (K & 0xFFFF) << 16;                       // tensor_dim0 lo16
      g1[2] = ((unsigned)K >> 16) | ((N & 0xFFFF) << 16); // dim0 hi / dim1 lo
      g1[3] = ((unsigned)N >> 16) | (BK << 16);         // dim1 hi / tile_dim0
      g1[4] = BN;                                       // tile_dim1 (tile_dim2=0)
      g1[5] = K;                                        // tensor_dim0_stride lo32
      g1[6] = 0;
      g1[7] = 0;
      v4i gz = {0, 0, 0, 0};
#if __has_include(<hip/amd_detail/amd_gfx1250_TDM.h>)
      v8i gz8 = {0, 0, 0, 0, 0, 0, 0, 0};
      __builtin_amdgcn_tensor_load_to_lds(g0, g1, gz, gz, gz8, 0);
#else
      __builtin_amdgcn_tensor_load_to_lds(g0, g1, gz, gz, 0);
#endif
      wait_tensorcnt0();
    }
#else
    for (int i = tid; i < (BN * BK) / 8; i += 128) {
      int r = i >> 2, q = i & 3;
      *(uint4*)&Bs[r * LDB + q * 8] =
          *(const uint4*)&Bw[(size_t)(n0 + r) * K + kt + q * 8];
    }
#endif
    if (kt + BK < K)  // prefetch next A tile (global_prefetch_b8)
      __builtin_prefetch(&A[(size_t)(m0 + tid) * K + kt + BK], 0, 1);
    __syncthreads();

    // ---- fragment loads per documented 16-bit layout (lane half-split over K)
    AFrag a0, a1, b[4];
    const int lhi = lane >> 4;
    const int mr  = lane & 15;
    const unsigned short* pa0 = &As[(wm * 32 + mr) * LDA];
    const unsigned short* pa1 = pa0 + 16 * LDA;
#pragma unroll
    for (int j = 0; j < 8; ++j) {
      int kb = ((j & 4) << 2) + lhi * 8 + ((j & 3) << 1);
      a0.u[2*j] = pa0[kb]; a0.u[2*j+1] = pa0[kb+1];
      a1.u[2*j] = pa1[kb]; a1.u[2*j+1] = pa1[kb+1];
#pragma unroll
      for (int ni = 0; ni < 4; ++ni) {
        const unsigned short* pb = &Bs[(ni * 16 + mr) * LDB];
        b[ni].u[2*j] = pb[kb]; b[ni].u[2*j+1] = pb[kb+1];
      }
    }
#pragma unroll
    for (int ni = 0; ni < 4; ++ni) {
      c[0][ni] = wmma_bf16(a0.v, b[ni].v, c[0][ni]);
      c[1][ni] = wmma_bf16(a1.v, b[ni].v, c[1][ni]);
    }
    __syncthreads();
  }

  // ---- epilogue: C/D layout -> VGPR r holds M=r (lanes<16) / 8+r, N=lane&15
  const int lhi = lane >> 4;
  const int nl  = lane & 15;
#pragma unroll
  for (int mi = 0; mi < 2; ++mi) {
#pragma unroll
    for (int ni = 0; ni < 4; ++ni) {
      int col = n0 + ni * 16 + nl;
      float bv = bias ? bias[col] : 0.f;
#pragma unroll
      for (int r = 0; r < 8; ++r) {
        int row = m0 + wm * 32 + mi * 16 + lhi * 8 + r;
        if (row < M) {
          float v = c[mi][ni][r] + bv;
          if (res)  v += res[(size_t)row * N + col];
          if (relu) v = fmaxf(v, 0.f);
          Cout[(size_t)row * N + col] = v;
        }
      }
    }
  }
}

// ---------------------------------------------------------------------------
// weight convert: out[n*K+k] (bf16) = trans ? W[k*N+n] : W[n*K+k]
// ---------------------------------------------------------------------------
__global__ void k_cvtw(const float* __restrict__ W, unsigned short* __restrict__ o,
                       int K, int N, int trans)
{
  size_t i = (size_t)blockIdx.x * blockDim.x + threadIdx.x;
  if (i >= (size_t)K * N) return;
  int k = (int)(i % K);
  int n = (int)(i / K);
  float v = trans ? W[(size_t)k * N + n] : W[(size_t)n * K + k];
  o[i] = f2bf(v);
}

// NCHW -> token-major [B*HW, C]
__global__ void k_nchw2tok(const float* __restrict__ x, float* __restrict__ y,
                           int B, int C, int HW)
{
  size_t i = (size_t)blockIdx.x * blockDim.x + threadIdx.x;
  if (i >= (size_t)B * HW * C) return;
  int c = (int)(i % C);
  size_t t = i / C;
  int p = (int)(t % HW);
  int b = (int)(t / HW);
  y[i] = x[((size_t)b * C + c) * HW + p];
}

// GroupNorm statistics: one block per (b, group), x token-major [B*HW, C]
__global__ __launch_bounds__(256) void k_gn_stats(const float* __restrict__ x,
                                                  float* __restrict__ mv,
                                                  int hw, int C, int G)
{
  int b = blockIdx.x / G, g = blockIdx.x % G;
  int cg = C / G;
  int n = hw * cg;
  float s = 0.f, s2 = 0.f;
  for (int i = threadIdx.x; i < n; i += 256) {
    int p = i / cg, c = i % cg;
    float v = x[((size_t)b * hw + p) * C + g * cg + c];
    s += v; s2 += v * v;
  }
  __shared__ float sh1[256], sh2[256];
  sh1[threadIdx.x] = s; sh2[threadIdx.x] = s2;
  __syncthreads();
  for (int o = 128; o > 0; o >>= 1) {
    if (threadIdx.x < o) { sh1[threadIdx.x] += sh1[threadIdx.x + o];
                           sh2[threadIdx.x] += sh2[threadIdx.x + o]; }
    __syncthreads();
  }
  if (threadIdx.x == 0) {
    float m = sh1[0] / n;
    float var = sh2[0] / n - m * m;
    mv[2 * blockIdx.x]     = m;
    mv[2 * blockIdx.x + 1] = rsqrtf(var + 1e-5f);
  }
}

__global__ void k_gn_apply_tok(const float* __restrict__ x, const float* __restrict__ mv,
                               const float* __restrict__ gam, const float* __restrict__ bet,
                               float* __restrict__ dst, int B, int hw, int C, int G,
                               int dstLen, int tokOff)
{
  size_t i = (size_t)blockIdx.x * blockDim.x + threadIdx.x;
  if (i >= (size_t)B * hw * C) return;
  int c = (int)(i % C);
  size_t tb = i / C;
  int p = (int)(tb % hw);
  int b = (int)(tb / hw);
  int g = c / (C / G);
  float m = mv[2 * (b * G + g)], r = mv[2 * (b * G + g) + 1];
  float v = (x[i] - m) * r * gam[c] + bet[c];
  dst[((size_t)b * dstLen + tokOff + p) * C + c] = v;
}

__global__ void k_gn_apply_nchw(const float* __restrict__ x, const float* __restrict__ mv,
                                const float* __restrict__ gam, const float* __restrict__ bet,
                                float* __restrict__ dst, int B, int hw, int C, int G)
{
  size_t i = (size_t)blockIdx.x * blockDim.x + threadIdx.x;
  if (i >= (size_t)B * hw * C) return;
  int c = (int)(i % C);
  size_t tb = i / C;
  int p = (int)(tb % hw);
  int b = (int)(tb / hw);
  int g = c / (C / G);
  float m = mv[2 * (b * G + g)], r = mv[2 * (b * G + g) + 1];
  float v = (x[i] - m) * r * gam[c] + bet[c];
  dst[((size_t)b * C + c) * hw + p] = v;
}

// sine positional embedding + level embed -> posf[2*hw, D]
__global__ void k_pos(float* __restrict__ posf, const float* __restrict__ lev,
                      int h, int w, int D)
{
  int i = blockIdx.x * blockDim.x + threadIdx.x;
  int hw = h * w;
  if (i >= 2 * hw * D) return;
  int d = i % D;
  int t = (i / D) % hw;
  int l = i / (D * hw);
  int nf = D / 2;
  int dd = d;
  float e;
  if (d < nf) { e = (float)(t / w + 1) / h * 6.28318530718f; }
  else        { dd = d - nf; e = (float)(t % w + 1) / w * 6.28318530718f; }
  int k = dd >> 1;
  float dim = __powf(10000.f, 2.f * k / (float)nf);
  float p = e / dim;
  float val = (dd & 1) ? __cosf(p) : __sinf(p);
  posf[i] = val + lev[l * D + d];
}

__global__ void k_add_pos(const float* __restrict__ src, const float* __restrict__ posf,
                          float* __restrict__ q, int B, int Len, int D)
{
  size_t i = (size_t)blockIdx.x * blockDim.x + threadIdx.x;
  if (i >= (size_t)B * Len * D) return;
  int d = (int)(i % D);
  int t = (int)((i / D) % Len);
  q[i] = src[i] + posf[(size_t)t * D + d];
}

// deformable bilinear sampling + softmax-weighted sum.
// One wave per (b, token, head); lane covers 2 of the 64 head channels.
__global__ void k_deform(const float* __restrict__ value, const float* __restrict__ off,
                         const float* __restrict__ awl, float* __restrict__ out,
                         int B, int hw, int h, int w)
{
  const int D = 512, Hd = 8, Dh = 64, L = 2, P = 4;
  int Len = 2 * hw;
  int gw = (int)(((size_t)blockIdx.x * blockDim.x + threadIdx.x) >> 5);
  int lane = threadIdx.x & 31;
  if (gw >= B * Len * Hd) return;
  int hd = gw % Hd;
  int t  = (gw / Hd) % Len;
  int b  = gw / (Hd * Len);
  int tt = t % hw;
  float rx = ((tt % w) + 0.5f) / w;
  float ry = ((tt / w) + 0.5f) / h;
  size_t bta = (size_t)(b * Len + t) * Hd + hd;
  const float* ofp = off + bta * (L * P * 2);
  const float* awp = awl + bta * (L * P);

  float lg[8], mx = -1e30f;
#pragma unroll
  for (int j = 0; j < 8; ++j) { lg[j] = awp[j]; mx = fmaxf(mx, lg[j]); }
  float ssum = 0.f;
#pragma unroll
  for (int j = 0; j < 8; ++j) { lg[j] = __expf(lg[j] - mx); ssum += lg[j]; }
  float inv = 1.f / ssum;

  int d0 = hd * Dh + lane;
  float acc0 = 0.f, acc1 = 0.f;
#pragma unroll
  for (int l = 0; l < L; ++l) {
#pragma unroll
    for (int p = 0; p < P; ++p) {
      int j = l * P + p;
      float lx = rx + ofp[2 * j]     / (float)w;
      float ly = ry + ofp[2 * j + 1] / (float)h;
      float x = lx * w - 0.5f, y = ly * h - 0.5f;
      float x0 = floorf(x), y0 = floorf(y);
      float wx = x - x0, wy = y - y0;
      int ix = (int)x0, iy = (int)y0;
      float wgt = lg[j] * inv;
      float cw[4] = {(1.f - wx) * (1.f - wy), wx * (1.f - wy),
                     (1.f - wx) * wy,         wx * wy};
      int cx[4] = {ix, ix + 1, ix, ix + 1};
      int cy[4] = {iy, iy, iy + 1, iy + 1};
#pragma unroll
      for (int c = 0; c < 4; ++c) {
        if (cx[c] >= 0 && cx[c] < w && cy[c] >= 0 && cy[c] < h) {
          size_t tok = (size_t)b * Len + l * hw + cy[c] * w + cx[c];
          const float* vp = value + tok * D + d0;
          float s = wgt * cw[c];
          acc0 += s * vp[0];
          acc1 += s * vp[32];
        }
      }
    }
  }
  float* op = out + ((size_t)(b * Len + t)) * D + d0;
  op[0]  = acc0;
  op[32] = acc1;
}

// modality-specific LayerNorm: one wave per token, params selected by token half
__global__ void k_ln(const float* __restrict__ x, const float* __restrict__ g2,
                     const float* __restrict__ b2, float* __restrict__ out,
                     int B, int Len, int n, int D)
{
  int gw = (int)(((size_t)blockIdx.x * blockDim.x + threadIdx.x) >> 5);
  int lane = threadIdx.x & 31;
  if (gw >= B * Len) return;
  int t = gw % Len;
  int sel = (t < n) ? 0 : 1;
  const float* g  = g2 + sel * D;
  const float* be = b2 + sel * D;
  const float* xp = x + (size_t)gw * D;
  float s = 0.f, s2 = 0.f;
  for (int d = lane; d < D; d += 32) { float v = xp[d]; s += v; s2 += v * v; }
  for (int o = 16; o > 0; o >>= 1) {
    s  += __shfl_xor(s,  o, 32);
    s2 += __shfl_xor(s2, o, 32);
  }
  float m = s / D;
  float r = rsqrtf(fmaxf(s2 / D - m * m, 0.f) + 1e-5f);
  float* op = out + (size_t)gw * D;
  for (int d = lane; d < D; d += 32) op[d] = (xp[d] - m) * r * g[d] + be[d];
}

// sum the two modality halves: out[b*hw+p, d] = src[b,p,d] + src[b,hw+p,d]
__global__ void k_sum2(const float* __restrict__ src, float* __restrict__ out,
                       int B, int hw, int D)
{
  size_t i = (size_t)blockIdx.x * blockDim.x + threadIdx.x;
  if (i >= (size_t)B * hw * D) return;
  int d = (int)(i % D);
  size_t bp = i / D;
  int p = (int)(bp % hw);
  int b = (int)(bp / hw);
  size_t s0 = ((size_t)b * 2 * hw + p) * D + d;
  out[i] = src[s0] + src[s0 + (size_t)hw * D];
}

// ---------------------------------------------------------------------------
// host orchestration
// ---------------------------------------------------------------------------
static inline int cdiv(long a, int b) { return (int)((a + b - 1) / b); }

static inline void gemm(hipStream_t s, const float* A, const unsigned short* Bw,
                        const float* bias, const float* res, float* C,
                        int M, int K, int N, int relu)
{
  dim3 g(N / BN, M / BM);
  k_gemm<<<g, 128, 0, s>>>(A, Bw, bias, res, C, M, K, N, relu);
}

static inline void cvtw(hipStream_t s, const float* W, unsigned short* o,
                        int K, int N, int trans)
{
  long n = (long)K * N;
  k_cvtw<<<cdiv(n, 256), 256, 0, s>>>(W, o, K, N, trans);
}

extern "C" void kernel_launch(void* const* d_in, const int* in_sizes, int n_in,
                              void* d_out, int out_size, void* d_ws, size_t ws_size,
                              hipStream_t stream)
{
  const int B = 16, Cc = 768, H = 20, Wd = 20, HW = 400, D = 512;
  const int Len = 2 * HW, NL = 6, Dff = 1024;
  const int M2 = B * Len;  // 12800
  const int M1 = B * HW;   // 6400

  const float* input_v = (const float*)d_in[0];
  const float* input_i = (const float*)d_in[1];
  const float* av_w  = (const float*)d_in[2];
  const float* av_b  = (const float*)d_in[3];
  const float* av_g  = (const float*)d_in[4];
  const float* av_be = (const float*)d_in[5];
  const float* ai_w  = (const float*)d_in[6];
  const float* ai_b  = (const float*)d_in[7];
  const float* ai_g  = (const float*)d_in[8];
  const float* ai_be = (const float*)d_in[9];
  const float* lev   = (const float*)d_in[10];
  const float* off_w = (const float*)d_in[11];
  const float* off_b = (const float*)d_in[12];
  const float* aw_w  = (const float*)d_in[13];
  const float* aw_b  = (const float*)d_in[14];
  const float* vp_w  = (const float*)d_in[15];
  const float* vp_b  = (const float*)d_in[16];
  const float* op_w  = (const float*)d_in[17];
  const float* op_b  = (const float*)d_in[18];
  const float* ln1_g = (const float*)d_in[19];
  const float* ln1_b = (const float*)d_in[20];
  const float* f1_w  = (const float*)d_in[21];
  const float* f1_b  = (const float*)d_in[22];
  const float* f2_w  = (const float*)d_in[23];
  const float* f2_b  = (const float*)d_in[24];
  const float* ln2_g = (const float*)d_in[25];
  const float* ln2_b = (const float*)d_in[26];
  const float* as_w  = (const float*)d_in[27];
  const float* as_b  = (const float*)d_in[28];
  const float* as_g  = (const float*)d_in[29];
  const float* as_be = (const float*)d_in[30];

  char* wp = (char*)d_ws;
  auto alloc = [&](size_t bytes) -> char* {
    char* p = wp;
    wp += (bytes + 255) & ~(size_t)255;
    return p;
  };
  float* src  = (float*)alloc((size_t)M2 * D * 4);
  float* qb   = (float*)alloc((size_t)M2 * D * 4);
  float* val  = (float*)alloc((size_t)M2 * D * 4);
  float* smp  = (float*)alloc((size_t)M2 * D * 4);
  float* ffh  = (float*)alloc((size_t)M2 * Dff * 4);
  float* offb = (float*)alloc((size_t)M2 * 128 * 4);
  float* awlb = (float*)alloc((size_t)M2 * 64 * 4);
  float* posf = (float*)alloc((size_t)Len * D * 4);
  unsigned short* wscr = (unsigned short*)alloc((size_t)Dff * D * 2);
  float* mv   = (float*)alloc((size_t)B * 32 * 2 * 4);
  float* tokv  = ffh;  // [M1, Cc] scratch during prologue/epilogue
  float* convo = smp;  // [M1, D]  scratch during prologue

  // ---- prologue: conv1x1 + GroupNorm for both modalities -> src ----
  k_nchw2tok<<<cdiv((long)M1 * Cc, 256), 256, 0, stream>>>(input_v, tokv, B, Cc, HW);
  cvtw(stream, av_w, wscr, Cc, D, 0);             // av_w is [D,Cc] = [N,K]
  gemm(stream, tokv, wscr, av_b, nullptr, convo, M1, Cc, D, 0);
  k_gn_stats<<<B * 32, 256, 0, stream>>>(convo, mv, HW, D, 32);
  k_gn_apply_tok<<<cdiv((long)M1 * D, 256), 256, 0, stream>>>(
      convo, mv, av_g, av_be, src, B, HW, D, 32, Len, 0);

  k_nchw2tok<<<cdiv((long)M1 * Cc, 256), 256, 0, stream>>>(input_i, tokv, B, Cc, HW);
  cvtw(stream, ai_w, wscr, Cc, D, 0);
  gemm(stream, tokv, wscr, ai_b, nullptr, convo, M1, Cc, D, 0);
  k_gn_stats<<<B * 32, 256, 0, stream>>>(convo, mv, HW, D, 32);
  k_gn_apply_tok<<<cdiv((long)M1 * D, 256), 256, 0, stream>>>(
      convo, mv, ai_g, ai_be, src, B, HW, D, 32, Len, HW);

  k_pos<<<cdiv((long)Len * D, 256), 256, 0, stream>>>(posf, lev, H, Wd, D);

  // ---- 6 encoder layers ----
  for (int i = 0; i < NL; ++i) {
    k_add_pos<<<cdiv((long)M2 * D, 256), 256, 0, stream>>>(src, posf, qb, B, Len, D);

    cvtw(stream, vp_w + (size_t)i * D * D, wscr, D, D, 1);
    gemm(stream, src, wscr, vp_b + i * D, nullptr, val, M2, D, D, 0);

    cvtw(stream, off_w + (size_t)i * D * 128, wscr, D, 128, 1);
    gemm(stream, qb, wscr, off_b + i * 128, nullptr, offb, M2, D, 128, 0);

    cvtw(stream, aw_w + (size_t)i * D * 64, wscr, D, 64, 1);
    gemm(stream, qb, wscr, aw_b + i * 64, nullptr, awlb, M2, D, 64, 0);

    k_deform<<<cdiv((long)B * Len * 8 * 32, 256), 256, 0, stream>>>(
        val, offb, awlb, smp, B, HW, H, Wd);

    cvtw(stream, op_w + (size_t)i * D * D, wscr, D, D, 1);
    gemm(stream, smp, wscr, op_b + i * D, src, qb, M2, D, D, 0);  // += residual
    k_ln<<<cdiv((long)M2 * 32, 256), 256, 0, stream>>>(
        qb, ln1_g + (size_t)i * 2 * D, ln1_b + (size_t)i * 2 * D, src, B, Len, HW, D);

    cvtw(stream, f1_w + (size_t)i * D * Dff, wscr, D, Dff, 1);
    gemm(stream, src, wscr, f1_b + i * Dff, nullptr, ffh, M2, D, Dff, 1);  // ReLU
    cvtw(stream, f2_w + (size_t)i * Dff * D, wscr, Dff, D, 1);
    gemm(stream, ffh, wscr, f2_b + i * D, src, qb, M2, Dff, D, 0);         // += residual
    k_ln<<<cdiv((long)M2 * 32, 256), 256, 0, stream>>>(
        qb, ln2_g + (size_t)i * 2 * D, ln2_b + (size_t)i * 2 * D, src, B, Len, HW, D);
  }

  // ---- epilogue: modality sum, conv1x1, GroupNorm -> d_out (NCHW) ----
  k_sum2<<<cdiv((long)M1 * D, 256), 256, 0, stream>>>(src, val, B, HW, D);
  cvtw(stream, as_w, wscr, D, Cc, 0);             // as_w is [Cc,D] = [N,K]
  gemm(stream, val, wscr, as_b, nullptr, ffh, M1, D, Cc, 0);
  k_gn_stats<<<B * 32, 256, 0, stream>>>(ffh, mv, HW, Cc, 32);
  k_gn_apply_nchw<<<cdiv((long)M1 * Cc, 256), 256, 0, stream>>>(
      ffh, mv, as_g, as_be, (float*)d_out, B, HW, Cc, 32);
}
